// BasicBlock_63496796504726
// MI455X (gfx1250) — matrix operand order repaired and checked by reading, not compiled, not run on hardware
//
#include <hip/hip_runtime.h>
#include <hip/hip_bf16.h>

// ---------------------------------------------------------------------------
// Problem constants (match reference)
// ---------------------------------------------------------------------------
#define NN 262144   // voxels
#define CC 128      // channels
#define KK 27       // kernel offsets
#define MM 131072   // map pairs per offset
#define TM 64       // M-rows per conv block

typedef __attribute__((ext_vector_type(16))) __bf16 v16bf;
typedef __attribute__((ext_vector_type(8)))  float  v8f;
typedef int b128i __attribute__((vector_size(16)));   // matches builtin param type

union Frag { v16bf v; int4 q[2]; };

// Async global->LDS gather path (CDNA5); falls back to plain copy if the
// builtins are unavailable (e.g. host pass).
#if __has_builtin(__builtin_amdgcn_global_load_async_to_lds_b128) && \
    __has_builtin(__builtin_amdgcn_s_wait_asynccnt)
#define ASYNC_GATHER 1
#else
#define ASYNC_GATHER 0
#endif

static __device__ __forceinline__ unsigned short f32_to_bf16(float f) {
    unsigned int u = __float_as_uint(f);
    unsigned int r = u + 0x7FFFu + ((u >> 16) & 1u);   // round-to-nearest-even
    return (unsigned short)(r >> 16);
}

// ---------------------------------------------------------------------------
// 1) prep: feats f32 -> bf16, zero f32 accumulator, zero BN stats
// ---------------------------------------------------------------------------
__global__ __launch_bounds__(256) void prep_kernel(
    const float* __restrict__ feats,
    unsigned short* __restrict__ featsb,
    float* __restrict__ acc,
    float* __restrict__ stats)
{
    size_t tid = (size_t)blockIdx.x * 256 + threadIdx.x;
    size_t i = tid * 4;
#pragma unroll
    for (int j = 0; j < 4; ++j) {
        featsb[i + j] = f32_to_bf16(feats[i + j]);
        acc[i + j] = 0.0f;
    }
    if (tid < 512) stats[tid] = 0.0f;   // stats1(sum,sq) + stats2(sum,sq)
}

// ---------------------------------------------------------------------------
// 2) wfrag: swizzle W[k][Cin][Cout] f32 into WMMA B-fragment order (bf16).
//    Fragment (k, nt, kc): 32 lanes x 16 bf16, element j of lane L:
//      N = nt*16 + (L&15),  K = kc*32 + (L>=16 ? 16 : 0) + j
// ---------------------------------------------------------------------------
__global__ __launch_bounds__(256) void wfrag_kernel(
    const float* __restrict__ W1, const float* __restrict__ W2,
    unsigned short* __restrict__ w1f, unsigned short* __restrict__ w2f)
{
    int tid = blockIdx.x * 256 + threadIdx.x;       // 2 * 27648 threads
    int which = tid / 27648;
    int rem   = tid % 27648;
    int lane = rem & 31;
    int kc   = (rem >> 5) & 3;
    int nt   = (rem >> 7) & 7;
    int k    = (rem >> 10);                          // 0..26
    const float* W = which ? W2 : W1;
    unsigned short* wf = which ? w2f : w1f;

    int n = nt * 16 + (lane & 15);
    int kbase = kc * 32 + ((lane >> 4) << 4);
    unsigned short tmp[16];
#pragma unroll
    for (int j = 0; j < 16; ++j)
        tmp[j] = f32_to_bf16(W[((size_t)k * CC + kbase + j) * CC + n]);

    unsigned short* dst = wf + ((size_t)((k * 8 + nt) * 4 + kc) * 32 + lane) * 16;
    *(int4*)(dst)     = *(const int4*)(tmp);
    *(int4*)(dst + 8) = *(const int4*)(tmp + 8);
}

// ---------------------------------------------------------------------------
// 3) conv: gather 64 rows -> LDS, 16x16x32 bf16 WMMA tiles, atomic scatter-add
//    grid = (MM/TM, KK), block = 256 (8 waves; wave w owns column tile w)
// ---------------------------------------------------------------------------
__global__ __launch_bounds__(256) void conv_kernel(
    const unsigned short* __restrict__ featsb,
    const unsigned short* __restrict__ wfrag,
    const int* __restrict__ in_maps,
    const int* __restrict__ out_maps,
    float* __restrict__ acc_out)
{
    __shared__ unsigned short As[TM * 136];   // 64 rows, padded stride 136 halves
    __shared__ int rowIn[TM];
    __shared__ int rowOut[TM];

    const int t  = threadIdx.x;
    const int k  = blockIdx.y;
    const int m0 = blockIdx.x * TM;

    if (t < TM)            rowIn[t]       = in_maps [k * MM + m0 + t];
    else if (t < 2 * TM)   rowOut[t - TM] = out_maps[k * MM + m0 + (t - TM)];
    __syncthreads();

    // Gather: 4 threads per row, 32 bf16 (64B) each -> 16KB tile in LDS.
    // 32-bit unsigned offsets keep the backend on the SADDR addressing form.
    {
        const int r  = t >> 2;
        const int ch = (t & 3) * 32;
        const unsigned goff = ((unsigned)rowIn[r] << 7) + (unsigned)ch;
        unsigned short* dst = As + r * 136 + ch;
#if ASYNC_GATHER
        const unsigned short* src = featsb + goff;
#pragma unroll
        for (int j = 0; j < 4; ++j)
            __builtin_amdgcn_global_load_async_to_lds_b128(
                (__attribute__((address_space(1))) b128i*)(uintptr_t)(src + j * 8),
                (__attribute__((address_space(3))) b128i*)(unsigned)(uintptr_t)(dst + j * 8),
                0, 0);
        __builtin_amdgcn_s_wait_asynccnt(0);
#else
        const int4* src = (const int4*)(featsb + goff);
        int4* d4 = (int4*)dst;
        d4[0] = src[0]; d4[1] = src[1]; d4[2] = src[2]; d4[3] = src[3];
#endif
    }
    __syncthreads();

    const int w    = t >> 5;       // wave -> column tile nt = w
    const int lane = t & 31;
    const int hi   = lane >> 4;    // 0: lanes 0-15, 1: lanes 16-31
    const int mrow = lane & 15;

    // Preload B fragments (4 K-chunks) straight from L2-resident swizzled W
    Frag bf[4];
    const unsigned short* wb = wfrag + (unsigned)((k * 8 + w) * 2048);
#pragma unroll
    for (int kc = 0; kc < 4; ++kc) {
        const int4* p = (const int4*)(wb + (unsigned)(kc * 512 + lane * 16));
        bf[kc].q[0] = p[0];
        bf[kc].q[1] = p[1];
    }

#pragma unroll
    for (int mt = 0; mt < 4; ++mt) {
        v8f c = {};
        const unsigned short* arow = As + (mt * 16 + mrow) * 136 + hi * 16;
#pragma unroll
        for (int kc = 0; kc < 4; ++kc) {
            Frag a;                                   // A 16x32 bf16 layout:
            a.q[0] = *(const int4*)(arow + kc * 32);      // K0..K0+7
            a.q[1] = *(const int4*)(arow + kc * 32 + 8);  // K0+8..K0+15
            c = __builtin_amdgcn_wmma_f32_16x16x32_bf16(
                    false, a.v, false, bf[kc].v, (short)0, c, false, false);
        }
        // Scatter D tile: VGPR i -> M = i (lanes 0-15) / 8+i (lanes 16-31)
        const int mb = mt * 16 + hi * 8;
        const unsigned cb = (unsigned)(w * 16 + (lane & 15));
#pragma unroll
        for (int i = 0; i < 8; ++i) {
            unsigned off = ((unsigned)rowOut[mb + i] << 7) + cb;
            atomicAdd(acc_out + off, c[i]);
        }
    }
}

// ---------------------------------------------------------------------------
// 4) bn_stats: per-channel sum / sumsq partials, atomics into stats[2*CC]
// ---------------------------------------------------------------------------
__global__ __launch_bounds__(128) void bn_stats_kernel(
    const float* __restrict__ x, float* __restrict__ stats, int rowsPerBlock)
{
    const int c = threadIdx.x;
    size_t r0 = (size_t)blockIdx.x * rowsPerBlock;
    float s = 0.0f, s2 = 0.0f;
    for (int r = 0; r < rowsPerBlock; ++r) {
        float v = x[(r0 + r) * CC + c];
        s += v; s2 += v * v;
    }
    atomicAdd(&stats[c], s);
    atomicAdd(&stats[CC + c], s2);
}

// ---------------------------------------------------------------------------
// 5) bn_apply: mid = relu(bn1(x)) -> bf16, and re-zero accumulator in place
// ---------------------------------------------------------------------------
__global__ __launch_bounds__(256) void bn_apply_kernel(
    float* __restrict__ x, const float* __restrict__ stats,
    const float* __restrict__ gamma, const float* __restrict__ beta,
    unsigned short* __restrict__ midb)
{
    size_t i = ((size_t)blockIdx.x * 256 + threadIdx.x) * 4;
    const float invN = 1.0f / (float)NN;
#pragma unroll
    for (int j = 0; j < 4; ++j) {
        int c = (int)((i + j) & (CC - 1));
        float mean = stats[c] * invN;
        float var  = stats[CC + c] * invN - mean * mean;
        float inv  = rsqrtf(var + 1e-5f);
        float y = (x[i + j] - mean) * inv * gamma[c] + beta[c];
        midb[i + j] = f32_to_bf16(fmaxf(y, 0.0f));
        x[i + j] = 0.0f;                 // accumulator ready for conv2
    }
}

// ---------------------------------------------------------------------------
// 6) final: out = relu(bn2(x) + feats)
// ---------------------------------------------------------------------------
__global__ __launch_bounds__(256) void final_kernel(
    const float* __restrict__ x, const float* __restrict__ stats,
    const float* __restrict__ gamma, const float* __restrict__ beta,
    const float* __restrict__ feats, float* __restrict__ out)
{
    size_t i = ((size_t)blockIdx.x * 256 + threadIdx.x) * 4;
    const float invN = 1.0f / (float)NN;
#pragma unroll
    for (int j = 0; j < 4; ++j) {
        int c = (int)((i + j) & (CC - 1));
        float mean = stats[c] * invN;
        float var  = stats[CC + c] * invN - mean * mean;
        float inv  = rsqrtf(var + 1e-5f);
        float y = (x[i + j] - mean) * inv * gamma[c] + beta[c] + feats[i + j];
        out[i + j] = fmaxf(y, 0.0f);
    }
}

// ---------------------------------------------------------------------------
// Launcher
// ---------------------------------------------------------------------------
extern "C" void kernel_launch(void* const* d_in, const int* in_sizes, int n_in,
                              void* d_out, int out_size, void* d_ws, size_t ws_size,
                              hipStream_t stream) {
    const float* feats    = (const float*)d_in[0];
    const int*   in_maps  = (const int*)  d_in[1];
    const int*   out_maps = (const int*)  d_in[2];
    const float* W1       = (const float*)d_in[3];
    const float* g1       = (const float*)d_in[4];
    const float* b1       = (const float*)d_in[5];
    const float* W2       = (const float*)d_in[6];
    const float* g2       = (const float*)d_in[7];
    const float* b2       = (const float*)d_in[8];
    float* out = (float*)d_out;

    const size_t NC = (size_t)NN * CC;
    const size_t WSZ = (size_t)KK * CC * CC;

    unsigned short* featsb = (unsigned short*)d_ws;          // NC bf16
    unsigned short* midb   = featsb + NC;                    // NC bf16
    float*          acc    = (float*)(midb + NC);            // NC f32
    unsigned short* w1f    = (unsigned short*)(acc + NC);    // KK*C*C bf16
    unsigned short* w2f    = w1f + WSZ;                      // KK*C*C bf16
    float*          stats  = (float*)(w2f + WSZ);            // 512 f32

    const int prepBlocks = (int)(NC / 1024);                 // 32768
    dim3 convGrid(MM / TM, KK);                              // (2048, 27)

    prep_kernel   <<<prepBlocks, 256, 0, stream>>>(feats, featsb, acc, stats);
    wfrag_kernel  <<<216, 256, 0, stream>>>(W1, W2, w1f, w2f);

    conv_kernel   <<<convGrid, 256, 0, stream>>>(featsb, w1f, in_maps, out_maps, acc);
    bn_stats_kernel<<<512, 128, 0, stream>>>(acc, stats, NN / 512);
    bn_apply_kernel<<<prepBlocks, 256, 0, stream>>>(acc, stats, g1, b1, midb);

    conv_kernel   <<<convGrid, 256, 0, stream>>>(midb, w2f, in_maps, out_maps, acc);
    bn_stats_kernel<<<512, 128, 0, stream>>>(acc, stats + 256, NN / 512);
    final_kernel  <<<prepBlocks, 256, 0, stream>>>(acc, stats + 256, g2, b2, feats, out);
}
